// BaselineModel_61246233641684
// MI455X (gfx1250) — matrix operand order repaired and checked by reading, not compiled, not run on hardware
//
#include <hip/hip_runtime.h>

// ---------------------------------------------------------------------------
// GCN forward on MI455X (gfx1250): wave32, WMMA bf16 -> f32 accumulate.
// Node state (25.6 MB f32 / 12.8 MB bf16) is L2-resident (192 MB L2), so the
// per-edge gather/scatter is L2 traffic, not HBM. GEMMs use
// v_wmma_f32_16x16x32_bf16 with B fragments register-resident and A fragments
// software-pipelined across grid-stride tiles.
// ---------------------------------------------------------------------------

typedef __bf16 v16bf __attribute__((ext_vector_type(16)));
typedef float  v8f   __attribute__((ext_vector_type(8)));

#define HDIM 128

// ---------------- edge prep ----------------

__global__ void prep_edges(const float* __restrict__ edge_attr,
                           const int* __restrict__ col,
                           float* __restrict__ ew,
                           float* __restrict__ deg, int E) {
    int e = blockIdx.x * blockDim.x + threadIdx.x;
    if (e >= E) return;
    float a = edge_attr[(size_t)e * 4 + 3];
    float w = 1.0f / (a * a);            // attr[:,3] ** -2
    ew[e] = w;
    atomicAdd(&deg[col[e]], w);          // global_atomic_add_f32 (no return)
}

__global__ void compute_dis(const float* __restrict__ deg,
                            float* __restrict__ dis, int N) {
    int i = blockIdx.x * blockDim.x + threadIdx.x;
    if (i >= N) return;
    dis[i] = rsqrtf(deg[i] + 1.0f);      // self-loop weight 1, deg+1 > 0 always
}

__global__ void compute_norm(const int* __restrict__ row,
                             const int* __restrict__ col,
                             const float* __restrict__ ew,
                             const float* __restrict__ dis,
                             float* __restrict__ norm, int E) {
    int e = blockIdx.x * blockDim.x + threadIdx.x;
    if (e >= E) return;
    norm[e] = dis[row[e]] * ew[e] * dis[col[e]];
}

// ---------------- weight convert f32 -> bf16 ----------------

__global__ void cvt_weights(const float* __restrict__ w0,
                            const float* __restrict__ w1,
                            const float* __restrict__ w2,
                            __bf16* __restrict__ o0,
                            __bf16* __restrict__ o1,
                            __bf16* __restrict__ o2) {
    int t = blockIdx.x * blockDim.x + threadIdx.x;
    if (t >= HDIM * HDIM) return;
    o0[t] = (__bf16)w0[t];
    o1[t] = (__bf16)w1[t];
    o2[t] = (__bf16)w2[t];
}

// ---------------- encoder layer 1: [N,16] x [16,128] + relu (tiny, VALU) ----

__global__ void encoder1(const float* __restrict__ x,
                         const float* __restrict__ xm,
                         const float* __restrict__ W,   // [16,128]
                         const float* __restrict__ b,   // [128]
                         __bf16* __restrict__ hb, int N) {
    int t = blockIdx.x * blockDim.x + threadIdx.x;
    if (t >= N * HDIM) return;
    int i = t >> 7, f = t & 127;
    float s = b[f];
    const float* xr  = x  + (size_t)i * 9;
    const float* xmr = xm + (size_t)i * 9;
#pragma unroll
    for (int j = 0; j < 8; ++j) s += xr[j]  * W[j * HDIM + f];
#pragma unroll
    for (int j = 0; j < 8; ++j) s += xmr[j] * W[(8 + j) * HDIM + f];
    s = fmaxf(s, 0.0f);
    hb[t] = (__bf16)s;
}

// ---------------- generic bf16 WMMA GEMM: [M,128] x [128,128] ----------------
// Block = 256 threads = 8 waves; each block owns a 16-row tile per grid-stride
// step, wave w owns output columns [16w, 16w+16). B fragments (4 k-chunks) are
// hoisted into registers for the whole kernel. A fragments for the next tile
// are prefetched while the current tile's 4 WMMAs run. In-place bf16 output is
// safe: a block reads only the 16 rows it later writes (prefetch reads rows of
// a strictly later, disjoint tile owned by the same block).
// Fragment layouts per CDNA5 ISA §7.12.2 (wave32):
//   A (16x32 bf16): lane L -> m = L%16; elem j: K = (j<8) ? 8*(L/16)+j
//                                                         : 16 + 8*(L/16)+(j-8)
//   B (32x16 bf16): lane L -> n = L%16; elem j: K same as A
//   C/D (16x16 f32): lane L -> n = L%16; elem r: m = 8*(L/16) + r

union AFrag { uint4 u[8]; v16bf v[4]; };

template <bool HAS_BIAS, bool RELU, bool OUT_F32, bool OUT_BF16>
__global__ void gemm_bf16_wmma(const __bf16* __restrict__ A,    // [M,128]
                               const __bf16* __restrict__ B,    // [128,128] (k,n)
                               const float* __restrict__ bias,  // [128] (if HAS_BIAS)
                               float* __restrict__ outF,        // [M,128] (if OUT_F32)
                               __bf16* __restrict__ outB,       // [M,128] (if OUT_BF16)
                               int M) {
    const int lane = threadIdx.x & 31;
    const int wave = threadIdx.x >> 5;          // 0..7
    const int n0   = wave * 16;
    const int half = lane >> 4;                 // 0 or 1
    const int ln   = lane & 15;

    // Hoist B fragments for this wave's 16-column tile (4 k-chunks of 32).
    v16bf bfrag[4];
#pragma unroll
    for (int kc = 0; kc < 4; ++kc) {
#pragma unroll
        for (int j = 0; j < 16; ++j) {
            int k = kc * 32 + half * 8 + (j & 7) + (j >> 3) * 16;
            bfrag[kc][j] = B[(size_t)k * HDIM + n0 + ln];
        }
    }
    float bval = 0.0f;
    if (HAS_BIAS) bval = bias[n0 + ln];

    // Per-lane base for A fragment loads: row = m0 + ln, two contiguous
    // 8-element (16 B) chunks per k-chunk at k = 32*kc + 8*half (+16).
    const __bf16* Abase = A + (size_t)ln * HDIM + half * 8;

    const int numTiles = M >> 4;                // M is a multiple of 16 (N=50000)
    int t = blockIdx.x;

    AFrag cur, nxt;
    if (t < numTiles) {
        const __bf16* ar = Abase + (size_t)t * 16 * HDIM;
#pragma unroll
        for (int kc = 0; kc < 4; ++kc) {
            cur.u[kc * 2]     = *reinterpret_cast<const uint4*>(ar + kc * 32);
            cur.u[kc * 2 + 1] = *reinterpret_cast<const uint4*>(ar + kc * 32 + 16);
        }
    }

    for (; t < numTiles; t += gridDim.x) {
        const int tn = t + gridDim.x;
        if (tn < numTiles) {                    // uniform branch: prefetch next tile
            const __bf16* ar = Abase + (size_t)tn * 16 * HDIM;
#pragma unroll
            for (int kc = 0; kc < 4; ++kc) {
                nxt.u[kc * 2]     = *reinterpret_cast<const uint4*>(ar + kc * 32);
                nxt.u[kc * 2 + 1] = *reinterpret_cast<const uint4*>(ar + kc * 32 + 16);
            }
        }

        v8f c = {0.f, 0.f, 0.f, 0.f, 0.f, 0.f, 0.f, 0.f};
#pragma unroll
        for (int kc = 0; kc < 4; ++kc)
            c = __builtin_amdgcn_wmma_f32_16x16x32_bf16(
                    false, cur.v[kc], false, bfrag[kc], (short)0, c, false, false);

        const int m0 = t * 16;
#pragma unroll
        for (int r = 0; r < 8; ++r) {
            float v = c[r];
            if (HAS_BIAS) v += bval;
            if (RELU) v = fmaxf(v, 0.0f);
            size_t off = (size_t)(m0 + half * 8 + r) * HDIM + n0 + ln;
            if (OUT_F32)  outF[off] = v;
            if (OUT_BF16) outB[off] = (__bf16)v;
        }

        if (tn < numTiles) cur = nxt;           // rotate double buffer
    }
}

// ---------------- per-edge scatter: agg[col] += norm * hw[row] --------------
// 32 threads (one wave) per edge, float4 per lane; hw/agg are L2-resident.

__global__ void edge_agg(const int* __restrict__ row,
                         const int* __restrict__ col,
                         const float* __restrict__ norm,
                         const float* __restrict__ hw,
                         float* __restrict__ agg, int E) {
    size_t tid = (size_t)blockIdx.x * blockDim.x + threadIdx.x;
    if (tid >= (size_t)E * 32) return;
    int e = (int)(tid >> 5);
    int g = ((int)tid & 31) << 2;               // feature offset 0..124 step 4
    float nm = norm[e];
    int r = row[e], c = col[e];
    const float4 v = *reinterpret_cast<const float4*>(hw + (size_t)r * HDIM + g);
    float* dst = agg + (size_t)c * HDIM + g;
    atomicAdd(dst + 0, nm * v.x);
    atomicAdd(dst + 1, nm * v.y);
    atomicAdd(dst + 2, nm * v.z);
    atomicAdd(dst + 3, nm * v.w);
}

// ---------------- finalize hop: + self-loop term + bias, ReLU, -> bf16 ------

__global__ void mp_finalize(const float* __restrict__ agg,
                            const float* __restrict__ hw,
                            const float* __restrict__ dis,
                            const float* __restrict__ b,
                            __bf16* __restrict__ hb, int N) {
    int t = blockIdx.x * blockDim.x + threadIdx.x;
    if (t >= N * HDIM) return;
    int i = t >> 7, f = t & 127;
    float d = dis[i];
    float v = agg[t] + d * d * hw[t] + b[f];
    hb[t] = (__bf16)fmaxf(v, 0.0f);
}

// ---------------- decoder layer 2: [N,128] x [128,3] (tiny, VALU) -----------

__global__ void decoder2(const __bf16* __restrict__ tIn,
                         const float* __restrict__ W,   // [128,3]
                         const float* __restrict__ b,   // [3]
                         float* __restrict__ out, int N) {
    int i = blockIdx.x * blockDim.x + threadIdx.x;
    if (i >= N) return;
    float s0 = b[0], s1 = b[1], s2 = b[2];
    const __bf16* r = tIn + (size_t)i * HDIM;
#pragma unroll 8
    for (int f = 0; f < HDIM; ++f) {
        float v = (float)r[f];
        s0 += v * W[f * 3 + 0];
        s1 += v * W[f * 3 + 1];
        s2 += v * W[f * 3 + 2];
    }
    out[(size_t)i * 3 + 0] = s0;
    out[(size_t)i * 3 + 1] = s1;
    out[(size_t)i * 3 + 2] = s2;
}

// ---------------------------------------------------------------------------

extern "C" void kernel_launch(void* const* d_in, const int* in_sizes, int n_in,
                              void* d_out, int out_size, void* d_ws, size_t ws_size,
                              hipStream_t stream) {
    const float* x     = (const float*)d_in[0];
    const float* xm    = (const float*)d_in[1];
    const int*   ei    = (const int*)d_in[2];
    const float* ea    = (const float*)d_in[3];
    // d_in[4] = batch (unused by forward)
    const float* Wenc1 = (const float*)d_in[5];
    const float* benc1 = (const float*)d_in[6];
    const float* Wenc2 = (const float*)d_in[7];
    const float* benc2 = (const float*)d_in[8];
    const float* Wgcn  = (const float*)d_in[9];
    const float* bgcn  = (const float*)d_in[10];
    const float* Wdec1 = (const float*)d_in[11];
    const float* bdec1 = (const float*)d_in[12];
    const float* Wdec2 = (const float*)d_in[13];
    const float* bdec2 = (const float*)d_in[14];
    float* out = (float*)d_out;

    const int N = in_sizes[0] / 9;      // 50000 (multiple of 16)
    const int E = in_sizes[2] / 2;      // 1,600,000
    const int* row = ei;                // edge_index[0]
    const int* col = ei + E;            // edge_index[1]

    // ---- carve workspace (256B aligned) ----
    char* p = (char*)d_ws;
    auto alloc = [&](size_t bytes) -> void* {
        void* r = (void*)p;
        p += (bytes + 255) & ~(size_t)255;
        return r;
    };
    float*  ew     = (float*)alloc((size_t)E * 4);
    float*  norm   = (float*)alloc((size_t)E * 4);
    float*  deg    = (float*)alloc((size_t)N * 4);
    float*  dis    = (float*)alloc((size_t)N * 4);
    float*  hw     = (float*)alloc((size_t)N * HDIM * 4);
    float*  agg    = (float*)alloc((size_t)N * HDIM * 4);
    __bf16* hb     = (__bf16*)alloc((size_t)N * HDIM * 2);
    __bf16* Wenc2b = (__bf16*)alloc((size_t)HDIM * HDIM * 2);
    __bf16* Wgcnb  = (__bf16*)alloc((size_t)HDIM * HDIM * 2);
    __bf16* Wdec1b = (__bf16*)alloc((size_t)HDIM * HDIM * 2);

    const int TPB = 256;
    const int GEMM_BLOCKS = 1024;       // grid-stride over 3125 tiles

    // ---- graph normalization precompute ----
    hipMemsetAsync(deg, 0, (size_t)N * 4, stream);
    cvt_weights<<<(HDIM * HDIM + TPB - 1) / TPB, TPB, 0, stream>>>(
        Wenc2, Wgcn, Wdec1, Wenc2b, Wgcnb, Wdec1b);
    prep_edges<<<(E + TPB - 1) / TPB, TPB, 0, stream>>>(ea, col, ew, deg, E);
    compute_dis<<<(N + TPB - 1) / TPB, TPB, 0, stream>>>(deg, dis, N);
    compute_norm<<<(E + TPB - 1) / TPB, TPB, 0, stream>>>(row, col, ew, dis, norm, E);

    // ---- encoder ----
    encoder1<<<(N * HDIM + TPB - 1) / TPB, TPB, 0, stream>>>(x, xm, Wenc1, benc1, hb, N);
    // enc2: h = h @ Wenc2 + b (no relu), in-place bf16
    gemm_bf16_wmma<true, false, false, true><<<GEMM_BLOCKS, TPB, 0, stream>>>(
        hb, Wenc2b, benc2, nullptr, hb, N);

    // ---- 4 message-passing hops (shared GCN weights) ----
    for (int s = 0; s < 4; ++s) {
        gemm_bf16_wmma<false, false, true, false><<<GEMM_BLOCKS, TPB, 0, stream>>>(
            hb, Wgcnb, nullptr, hw, nullptr, N);
        hipMemsetAsync(agg, 0, (size_t)N * HDIM * 4, stream);
        size_t ethreads = (size_t)E * 32;
        edge_agg<<<(unsigned)((ethreads + TPB - 1) / TPB), TPB, 0, stream>>>(
            row, col, norm, hw, agg, E);
        mp_finalize<<<(N * HDIM + TPB - 1) / TPB, TPB, 0, stream>>>(
            agg, hw, dis, bgcn, hb, N);
    }

    // ---- decoder ----
    gemm_bf16_wmma<true, true, false, true><<<GEMM_BLOCKS, TPB, 0, stream>>>(
        hb, Wdec1b, bdec1, nullptr, hb, N);
    decoder2<<<(N + TPB - 1) / TPB, TPB, 0, stream>>>(hb, Wdec2, bdec2, out, N);
}
